// GeneralAttention_82927228551221
// MI455X (gfx1250) — matrix operand order repaired and checked by reading, not compiled, not run on hardware
//
#include <hip/hip_runtime.h>
#include <hip/hip_bf16.h>

// CDNA5 / gfx1250: wave32, v_wmma_f32_16x16x32_f16 (f32 accumulate).
typedef __attribute__((ext_vector_type(16))) _Float16 v16h;
typedef __attribute__((ext_vector_type(8)))  _Float16 v8h;
typedef __attribute__((ext_vector_type(4)))  _Float16 v4h;
typedef __attribute__((ext_vector_type(8)))  float    v8f;
typedef __attribute__((ext_vector_type(4)))  float    f4;

#define BM       128
#define BN       128
#define BK       64
#define LDK      (BK + 8)   // +16B row pad to spread LDS banks
#define NTHREADS 256        // 8 wave32s

// ---------------------------------------------------------------------------
// Tiled WMMA GEMM:  C[M,N] = A[M,K] * op(B)   (+ optional scale on store)
//   TA   : A element type in memory (float -> cvt f16, _Float16 -> copy)
//   B_NT : true  -> B stored [N,K] row-major ("NT", k-contiguous)
//          false -> B stored [K,N] row-major ("NN", transposed while staging)
//   TC   : output element type
// Block tile 128x128, K-step 64. 8 waves: wave (wm,wn) owns a 32x64 strip
// (2 x 4 accumulators). Batched via blockIdx.z with explicit strides.
// ---------------------------------------------------------------------------
template <typename TA, bool B_NT, typename TC, bool DO_SCALE>
__global__ __launch_bounds__(NTHREADS)
void gemm_wmma_kernel(const TA* __restrict__ Ag, long long sA, int lda,
                      const float* __restrict__ Bg, long long sB, int ldb,
                      TC* __restrict__ Cg, long long sC, int ldc,
                      int M, int N, int K, float scale)
{
    __shared__ _Float16 As[BM][LDK];   // [m][k]
    __shared__ _Float16 Bs[BN][LDK];   // [n][k]  (always k-contiguous)

    const int tid  = threadIdx.x;
    const int lane = tid & 31;
    const int wave = tid >> 5;
    const int wm   = wave & 3;    // 0..3 : 32-row strip
    const int wn   = wave >> 2;   // 0..1 : 64-col strip

    const int m0 = blockIdx.x * BM;
    const int n0 = blockIdx.y * BN;

    const TA*    A = Ag + (long long)blockIdx.z * sA;
    const float* B = Bg + (long long)blockIdx.z * sB;
    TC*          C = Cg + (long long)blockIdx.z * sC;

    v8f acc[2][4];
#pragma unroll
    for (int mi = 0; mi < 2; ++mi)
#pragma unroll
        for (int ni = 0; ni < 4; ++ni)
#pragma unroll
            for (int i = 0; i < 8; ++i) acc[mi][ni][i] = 0.0f;

    // Staging geometry: 128x64 tile = 8192 elems, 32 per thread
    const int srow = tid >> 1;          // 0..127
    const int scol = (tid & 1) * 32;    // 0 or 32
    // NN path: values tile [64][128], 32 floats per thread along a k-row
    const int vkr  = tid >> 2;          // 0..63
    const int vc0  = (tid & 3) * 32;    // 0,32,64,96

    for (int k0 = 0; k0 < K; k0 += BK) {
        // ---- stage A tile (128x64) -> f16 LDS ----
        {
            const TA* aptr = &A[(long long)(m0 + srow) * lda + k0 + scol];
            if (sizeof(TA) == 4) {                 // fp32 -> f16
                const float* af = (const float*)aptr;
#pragma unroll
                for (int j = 0; j < 32; j += 4) {
                    f4 x = *(const f4*)(af + j);
                    v4h h;
#pragma unroll
                    for (int i = 0; i < 4; ++i) h[i] = (_Float16)x[i];
                    *(v4h*)&As[srow][scol + j] = h;
                }
            } else {                               // already f16
                const _Float16* ah = (const _Float16*)aptr;
#pragma unroll
                for (int j = 0; j < 32; j += 8)
                    *(v8h*)&As[srow][scol + j] = *(const v8h*)(ah + j);
            }
            if (k0 + BK < K)
                __builtin_prefetch(&A[(long long)(m0 + srow) * lda + k0 + BK + scol], 0, 1);
        }
        // ---- stage B tile -> f16 LDS, layout [n][k] ----
        if (B_NT) {
            const float* bptr = &B[(long long)(n0 + srow) * ldb + k0 + scol];
#pragma unroll
            for (int j = 0; j < 32; j += 4) {
                f4 x = *(const f4*)(bptr + j);
                v4h h;
#pragma unroll
                for (int i = 0; i < 4; ++i) h[i] = (_Float16)x[i];
                *(v4h*)&Bs[srow][scol + j] = h;
            }
            if (k0 + BK < K)
                __builtin_prefetch(&B[(long long)(n0 + srow) * ldb + k0 + BK + scol], 0, 1);
        } else {
            // B is [K,N]; read 32 contiguous floats of a k-row, transpose into Bs
            const float* bptr = &B[(long long)(k0 + vkr) * ldb + n0 + vc0];
#pragma unroll
            for (int j = 0; j < 32; j += 4) {
                f4 x = *(const f4*)(bptr + j);
#pragma unroll
                for (int i = 0; i < 4; ++i) Bs[vc0 + j + i][vkr] = (_Float16)x[i];
            }
            if (k0 + BK < K)
                __builtin_prefetch(&B[(long long)(k0 + BK + vkr) * ldb + n0 + vc0], 0, 1);
        }
        __syncthreads();

        // ---- 2 K-substeps of 32 within the staged 64 ----
        const int khalf = lane >> 4;
        const int lrow  = lane & 15;
#pragma unroll
        for (int ks = 0; ks < BK; ks += 32) {
            // A fragments: lanes 0-15 K{0-7,16-23}, lanes 16-31 K{8-15,24-31}
            v16h afrag[2];
#pragma unroll
            for (int mi = 0; mi < 2; ++mi) {
                const _Float16* arp = &As[wm * 32 + mi * 16 + lrow][ks];
                v8h lo = *(const v8h*)(arp + khalf * 8);
                v8h hi = *(const v8h*)(arp + 16 + khalf * 8);
#pragma unroll
                for (int i = 0; i < 8; ++i) { afrag[mi][i] = lo[i]; afrag[mi][8 + i] = hi[i]; }
            }
            // B fragments: lane n = column n; lanes 0-15 K{0-7,16-23}, lanes 16-31 K{8-15,24-31}
#pragma unroll
            for (int ni = 0; ni < 4; ++ni) {
                const _Float16* brp = &Bs[wn * 64 + ni * 16 + lrow][ks + khalf * 8];
                v8h blo = *(const v8h*)(brp);
                v8h bhi = *(const v8h*)(brp + 16);
                v16h bfrag;
#pragma unroll
                for (int i = 0; i < 8; ++i) { bfrag[i] = blo[i]; bfrag[8 + i] = bhi[i]; }
#pragma unroll
                for (int mi = 0; mi < 2; ++mi)
                    acc[mi][ni] = __builtin_amdgcn_wmma_f32_16x16x32_f16(
                        false, afrag[mi], false, bfrag, (short)0, acc[mi][ni], false, false);
            }
        }
        __syncthreads();
    }

    // ---- store C: lane holds column (lane&15); VGPR r -> M = r + 8*(lane>>4)
    const int cn = lane & 15;
    const int mh = (lane >> 4) * 8;
#pragma unroll
    for (int mi = 0; mi < 2; ++mi)
#pragma unroll
        for (int ni = 0; ni < 4; ++ni) {
            const int col = n0 + wn * 64 + ni * 16 + cn;
#pragma unroll
            for (int r = 0; r < 8; ++r) {
                const int row = m0 + wm * 32 + mi * 16 + mh + r;
                float v = acc[mi][ni][r];
                if (DO_SCALE) v *= scale;
                C[(long long)row * ldc + col] = (TC)v;
            }
        }
}

// ---------------------------------------------------------------------------
// In-place masked softmax over each length-S row of attn (rows = B*Q).
// Matches reference: masked -> -inf before softmax => attn = 0 there.
// attn (134 MB) fits in the 192 MB L2, so these passes are L2-resident.
// ---------------------------------------------------------------------------
__global__ __launch_bounds__(NTHREADS)
void masked_softmax_kernel(float* __restrict__ attn,
                           const unsigned char* __restrict__ mask,
                           int S, int Q)
{
    __shared__ float red[NTHREADS];
    const int row = blockIdx.x;
    const int b   = row / Q;
    float* p = attn + (long long)row * S;
    const unsigned char* m = mask + (long long)b * S;
    const int tid = threadIdx.x;

    float mx = -3.402823466e38f;
    for (int i = tid; i < S; i += NTHREADS)
        if (m[i]) mx = fmaxf(mx, p[i]);
    red[tid] = mx; __syncthreads();
    for (int s = NTHREADS / 2; s > 0; s >>= 1) {
        if (tid < s) red[tid] = fmaxf(red[tid], red[tid + s]);
        __syncthreads();
    }
    mx = red[0];
    __syncthreads();

    float sum = 0.0f;
    for (int i = tid; i < S; i += NTHREADS) {
        float e = m[i] ? __expf(p[i] - mx) : 0.0f;
        p[i] = e;
        sum += e;
    }
    red[tid] = sum; __syncthreads();
    for (int s = NTHREADS / 2; s > 0; s >>= 1) {
        if (tid < s) red[tid] += red[tid + s];
        __syncthreads();
    }
    const float inv = 1.0f / red[0];
    for (int i = tid; i < S; i += NTHREADS)
        p[i] *= inv;
}

// ---------------------------------------------------------------------------
extern "C" void kernel_launch(void* const* d_in, const int* in_sizes, int n_in,
                              void* d_out, int out_size, void* d_ws, size_t ws_size,
                              hipStream_t stream)
{
    (void)in_sizes; (void)n_in; (void)out_size; (void)ws_size;
    const int B = 64, Q = 128, S = 4096, H = 1024;

    const float* query  = (const float*)d_in[0];
    const float* keys   = (const float*)d_in[1];
    const float* values = (const float*)d_in[2];
    const unsigned char* mask = (const unsigned char*)d_in[3]; // jnp.bool_ = 1 byte
    const float* Wa     = (const float*)d_in[4];

    float* ctx  = (float*)d_out;                               // [B,Q,H]
    float* attn = ctx + (long long)B * Q * H;                  // [B,Q,S]
    _Float16* qp = (_Float16*)d_ws;                            // [B*Q, H] f16 (16 MiB)

    dim3 blk(NTHREADS);

    // 1) qp = query @ Wa^T   ([8192,1024] = [8192,1024] x NT[1024,1024])
    gemm_wmma_kernel<float, true, _Float16, false>
        <<<dim3((B * Q) / BM, H / BN, 1), blk, 0, stream>>>(
            query, 0, H, Wa, 0, H, qp, 0, H, B * Q, H, H, 1.0f);

    // 2) scores = (qp @ keys^T) / sqrt(H), per batch ([128,4096])
    gemm_wmma_kernel<_Float16, true, float, true>
        <<<dim3(Q / BM, S / BN, B), blk, 0, stream>>>(
            qp, (long long)Q * H, H, keys, (long long)S * H, H,
            attn, (long long)Q * S, S, Q, S, H, 0.03125f);

    // 3) masked softmax in place over attn rows
    masked_softmax_kernel<<<dim3(B * Q), blk, 0, stream>>>(attn, mask, S, Q);

    // 4) ctx = attn @ values, per batch ([128,1024] = [128,4096] x NN[4096,1024])
    gemm_wmma_kernel<float, false, float, false>
        <<<dim3(Q / BM, H / BN, B), blk, 0, stream>>>(
            attn, (long long)Q * S, S, values, (long long)S * H, H,
            ctx, (long long)Q * H, H, Q, H, S, 1.0f);
}
